// DeformConv_69217692942559
// MI455X (gfx1250) — matrix operand order repaired and checked by reading, not compiled, not run on hardware
//
#include <hip/hip_runtime.h>

// Problem constants (match the reference)
#define Bn   4
#define Cn   64
#define Hn   128
#define Wn   128
#define On   64
#define KKn  9           // 3x3 taps
#define KTOT (Cn * KKn)  // 576 = GEMM K
#define HWn  (Hn * Wn)
#define KSTEPS (KTOT / 32)     // 18 WMMA K-steps

#define TILE_P 32              // pixels (N) per block
#define SLOTS  (KKn * TILE_P)  // 288 coefficient slots per block
#define MT_ROW 592             // mappedT row stride in bf16 (576 + 16 pad; 1184 B = 37*32 -> 32B aligned rows)

typedef __attribute__((ext_vector_type(16))) __bf16 v16bf;
typedef __attribute__((ext_vector_type(8)))  float  v8f;

// ---- kernel 1: weight f32 -> bf16, pre-packed into WMMA A-fragment order ----
// Packed index: (((kstep*64 + m)*2 + half)*16 + i), where k = kstep*32 + half*16 + i
// so each (lane m, half) reads its full 16-bf16 A fragment as one contiguous 32B chunk.
__global__ void wpack_kernel(const float* __restrict__ w, __bf16* __restrict__ wp, int n) {
    int e = blockIdx.x * 256 + threadIdx.x;
    if (e >= n) return;
    const int i16   = e & 15;
    const int half  = (e >> 4) & 1;
    const int m     = (e >> 5) & 63;
    const int kstep = e >> 11;
    const int k = kstep * 32 + half * 16 + i16; // contiguous: k = half*16 + i
    wp[e] = (__bf16)w[m * KTOT + k];
}

// ---- kernel 2: deformable conv: LDS im2col (bf16) + WMMA GEMM ----
__global__ __launch_bounds__(256) void deform_conv_wmma(
    const float* __restrict__ x,       // [B,C,H,W]
    const float* __restrict__ off,     // [B,2*KK,H,W]
    const __bf16* __restrict__ wp,     // packed A fragments [18][64][2][16] bf16
    float* __restrict__ out)           // [B,O,H,W]
{
    __shared__ int    sa[4][SLOTS];                         // clamped plane offsets (4 corners)
    __shared__ float  sw[4][SLOTS];                         // mask-folded bilinear weights
    __shared__ __align__(32) __bf16 mt[TILE_P * MT_ROW];    // mappedT[pixel][k] bf16

    const int tid = threadIdx.x;
    const int w0  = blockIdx.x * TILE_P;
    const int h   = blockIdx.y;
    const int b   = blockIdx.z;

    // ---- phase 1a: bilinear coefficients, one slot per (tap, pixel) ----
    for (int s = tid; s < SLOTS; s += 256) {
        const int t = s >> 5;          // tap 0..8
        const int p = s & 31;          // pixel in tile
        const int w = w0 + p;
        const float oi = off[(((b * 2 * KKn) + 2 * t    ) * Hn + h) * Wn + w];
        const float oj = off[(((b * 2 * KKn) + 2 * t + 1) * Hn + h) * Wn + w];
        const float ci = oi + (float)(h + t / 3 - 1);
        const float cj = oj + (float)(w + t % 3 - 1);
        const float fi0 = floorf(ci), fj0 = floorf(cj);
        const int i0 = (int)fi0, j0 = (int)fj0;
        const int i1 = i0 + 1,   j1 = j0 + 1;
        const float fi = ci - fi0, fj = cj - fj0;
        const float mi0 = (i0 >= 0 && i0 < Hn) ? 1.f : 0.f;
        const float mi1 = (i1 >= 0 && i1 < Hn) ? 1.f : 0.f;
        const float mj0 = (j0 >= 0 && j0 < Wn) ? 1.f : 0.f;
        const float mj1 = (j1 >= 0 && j1 < Wn) ? 1.f : 0.f;
        const int ic0 = min(max(i0, 0), Hn - 1), ic1 = min(max(i1, 0), Hn - 1);
        const int jc0 = min(max(j0, 0), Wn - 1), jc1 = min(max(j1, 0), Wn - 1);
        sa[0][s] = ic0 * Wn + jc0;  sw[0][s] = (1.f - fi) * (1.f - fj) * mi0 * mj0;  // lt
        sa[1][s] = ic0 * Wn + jc1;  sw[1][s] = (1.f - fi) * fj         * mi0 * mj1;  // rt
        sa[2][s] = ic1 * Wn + jc0;  sw[2][s] = fi         * (1.f - fj) * mi1 * mj0;  // lb
        sa[3][s] = ic1 * Wn + jc1;  sw[3][s] = fi         * fj         * mi1 * mj1;  // rb
    }
    __syncthreads();

    // ---- phase 1b: gather + blend all channels into LDS im2col tile (bf16) ----
    const float* __restrict__ xb = x + (size_t)b * Cn * HWn;
    for (int e = tid; e < Cn * SLOTS; e += 256) {
        const int c = e / SLOTS;
        const int s = e - c * SLOTS;
        const float* __restrict__ px = xb + c * HWn;
        const float v = sw[0][s] * px[sa[0][s]] + sw[1][s] * px[sa[1][s]]
                      + sw[2][s] * px[sa[2][s]] + sw[3][s] * px[sa[3][s]];
        const int t = s >> 5, p = s & 31;
        mt[p * MT_ROW + c * KKn + t] = (__bf16)v;   // k = c*9 + t matches weight [o][c][tap]
    }
    __syncthreads();

    // ---- phase 2: per-wave 16x16 WMMA GEMM tiles ----
    const int wave  = tid >> 5, lane = tid & 31;
    const int mtile = wave & 3;       // 4 tiles of 16 output channels
    const int ntile = wave >> 2;      // 2 tiles of 16 pixels
    const int half  = lane >> 4, l15 = lane & 15;

    // Packed A: this lane's fragment for kstep ks lives at ((ks*64 + m)*2 + half)*16
    const __bf16* __restrict__ afrag = wp + (size_t)(((mtile * 16 + l15) * 2 + half) * 16);
    const __bf16* __restrict__ brow  = mt + (ntile * 16 + l15) * MT_ROW;

    v8f acc = {};
#pragma unroll
    for (int ks = 0; ks < KSTEPS; ++ks) {
        // A fragment: one aligned 32B read, already in VGPR order (no packing VALU)
        const v16bf A = *(const v16bf*)(afrag + (size_t)ks * (64 * 2 * 16));
        // B fragment (32x16 bf16): lanes 0-15 cover K 0..15, lanes 16-31 cover K 16..31 -> one 32B read
        const v16bf Bf = *(const v16bf*)(brow + ks * 32 + half * 16);
        acc = __builtin_amdgcn_wmma_f32_16x16x32_bf16(
            /*neg_a=*/false, A, /*neg_b=*/false, Bf,
            /*c_mod=*/(short)0, acc, /*reuse_a=*/false, /*reuse_b=*/false);
    }

    // ---- store: C/D layout: VGPR r -> M = r + half*8, N = lane&15 ----
    const int wpix = w0 + ntile * 16 + l15;
#pragma unroll
    for (int r = 0; r < 8; ++r) {
        const int o = mtile * 16 + half * 8 + r;
        out[(((size_t)b * On + o) * Hn + h) * Wn + wpix] = acc[r];
    }
}

extern "C" void kernel_launch(void* const* d_in, const int* in_sizes, int n_in,
                              void* d_out, int out_size, void* d_ws, size_t ws_size,
                              hipStream_t stream) {
    const float* x   = (const float*)d_in[0];   // [4,64,128,128]
    const float* off = (const float*)d_in[1];   // [4,18,128,128]
    const float* wgt = (const float*)d_in[2];   // [64,64,3,3]
    float* out = (float*)d_out;                 // [4,64,128,128]
    __bf16* wp = (__bf16*)d_ws;                 // 18*64*2*16 bf16 = 72 KB scratch

    const int nw = On * KTOT;                   // 36864 packed elements
    wpack_kernel<<<(nw + 255) / 256, 256, 0, stream>>>(wgt, wp, nw);

    dim3 grid(Wn / TILE_P, Hn, Bn);             // 4 x 128 x 4 = 2048 blocks
    deform_conv_wmma<<<grid, 256, 0, stream>>>(x, off, wp, out);
}
